// BitConv2d_16569983828612
// MI455X (gfx1250) — 8-bit matrix operand order repaired and checked by reading, compiled, not run on hardware
//
#include <hip/hip_runtime.h>
#include <stdint.h>


#define NB    32
#define CIN   256
#define HH    56
#define WW    56
#define HW    (HH*WW)
#define COUT  256
#define NPIX  (NB*HW)
#define KW    2304
#define GEPS  1e-5f
#define QEPS  1e-8f
#define ROWP  272

typedef int v8i __attribute__((ext_vector_type(8)));

__global__ __launch_bounds__(256) void gn_stats_kernel(
    const float* __restrict__ x, float* __restrict__ mu, float* __restrict__ rs) {
  const int bg  = blockIdx.x;
  const int tid = threadIdx.x;
  const float* p = x + (size_t)bg * (32 * HW);
  float s = 0.f, s2 = 0.f;
  for (int i = tid; i < 32 * HW; i += 256) {
    float v = p[i];
    s += v; s2 += v * v;
  }
  __shared__ float shA[256];
  __shared__ float shB[256];
  shA[tid] = s; shB[tid] = s2;
  __syncthreads();
  for (int off = 128; off > 0; off >>= 1) {
    if (tid < off) { shA[tid] += shA[tid + off]; shB[tid] += shB[tid + off]; }
    __syncthreads();
  }
  if (tid == 0) {
    const float inv_n = 1.0f / (float)(32 * HW);
    float m   = shA[0] * inv_n;
    float var = shB[0] * inv_n - m * m;
    mu[bg] = m;
    rs[bg] = rsqrtf(var + GEPS);
  }
}

__global__ __launch_bounds__(256) void act_quant_kernel(
    const float* __restrict__ x, const float* __restrict__ gamma,
    const float* __restrict__ beta, const float* __restrict__ a,
    const float* __restrict__ mu, const float* __restrict__ rs,
    unsigned char* __restrict__ actQ) {
  const int t   = blockIdx.x * 256 + threadIdx.x;
  const int pix = t >> 6;
  const int c4  = (t & 63) << 2;
  const int b   = pix / HW;
  const int hw  = pix - b * HW;
  const float ac = fmaxf(a[0], 0.f) + QEPS;
  const float S  = 255.f / ac;
  const int   g  = c4 >> 5;
  const float m  = mu[b * 8 + g];
  const float r  = rs[b * 8 + g];
  unsigned int packed = 0u;
#pragma unroll
  for (int j = 0; j < 4; ++j) {
    const int c = c4 + j;
    float v  = x[((size_t)(b * CIN + c)) * HW + hw];
    float xn = (v - m) * r * gamma[c] + beta[c];
    float h  = fmaxf(xn, 0.f);
    h = h * h;
    h = fminf(h, ac);
    unsigned int q = (unsigned int)(int)rintf(h * S);
    packed |= (q & 255u) << (8 * j);
  }
  *(unsigned int*)(actQ + (size_t)pix * CIN + c4) = packed;
}

__global__ __launch_bounds__(256) void wquant_kernel(
    const float* __restrict__ wfp, const float* __restrict__ a,
    signed char* __restrict__ wqs, float* __restrict__ scale) {
  const int oc  = blockIdx.x;
  const int tid = threadIdx.x;
  const float* w = wfp + (size_t)oc * KW;
  __shared__ float shA[256];
  __shared__ float shB[256];
  __shared__ float tsh, ash;

  float sa = 0.f;
  for (int i = tid; i < KW; i += 256) sa += fabsf(w[i]);
  shA[tid] = sa; __syncthreads();
  for (int off = 128; off > 0; off >>= 1) {
    if (tid < off) shA[tid] += shA[tid + off];
    __syncthreads();
  }
  if (tid == 0) tsh = 0.05f * shA[0] / (float)KW;
  __syncthreads();
  const float thr = tsh;
  __syncthreads();

  float swa = 0.f, cnt = 0.f;
  for (int i = tid; i < KW; i += 256) {
    float aw = fabsf(w[i]);
    if (aw > thr) { swa += aw; cnt += 1.f; }
  }
  shA[tid] = swa; shB[tid] = cnt; __syncthreads();
  for (int off = 128; off > 0; off >>= 1) {
    if (tid < off) { shA[tid] += shA[tid + off]; shB[tid] += shB[tid + off]; }
    __syncthreads();
  }
  if (tid == 0) ash = shA[0] / (shB[0] + QEPS);

  const int mt = oc >> 4;
  const int lr = oc & 15;
  for (int i = tid; i < KW; i += 256) {
    const int ci = i / 9;
    const int kp = i - ci * 9;
    const float v = w[i];
    signed char s = 0;
    if (fabsf(v) > thr) s = (v > 0.f) ? (signed char)1 : (signed char)-1;
    const int kci  = ci >> 6;
    const int koff = ci & 63;
    const int j    = ((koff >> 5) << 1) | ((koff >> 3) & 1);
    const int hi2  = (koff >> 4) & 1;
    const int o    = koff & 7;
    const int lane = hi2 * 16 + lr;
    const size_t idx =
        ((((size_t)kp * 16 + mt) * 4 + kci) * 32 + lane) * 32 + j * 8 + o;
    wqs[idx] = s;
  }
  if (tid == 0) {
    const float ac = fmaxf(a[0], 0.f) + QEPS;
    scale[oc] = ash * ac / 255.f;
  }
}

__global__ __launch_bounds__(256) void conv_gemm_kernel(
    const unsigned char* __restrict__ actQ,
    const signed char*  __restrict__ wqs,
    const float* __restrict__ scale, const float* __restrict__ bias,
    float* __restrict__ out) {
  __shared__ __align__(16) unsigned char stage[2][16 * ROWP];
  __shared__ __align__(16) unsigned char zbuf[16];

  const int tid  = threadIdx.x;
  const int lane = tid & 31;
  const int wv   = tid >> 5;
  const int hi   = lane >> 4;
  const int lr   = lane & 15;
  const int n0   = blockIdx.x * 16;

  const int sp  = tid >> 4;
  const int sc  = (tid & 15) << 4;
  const int ns  = n0 + sp;
  const int bs  = ns / HW;
  const int rsd = ns - bs * HW;
  const int hs  = rsd / WW;
  const int wsd = rsd - hs * WW;

  const int nc = n0 + lr;
  const int bc = nc / HW;
  const int rc = nc - bc * HW;
  const int hc = rc / WW;
  const int wc = rc - hc * WW;

  if (tid == 0) *(uint4*)zbuf = make_uint4(0u, 0u, 0u, 0u);

  const int m0  = wv * 32;
  const int mt0 = wv * 2;
  const unsigned lds0 = (unsigned)(uintptr_t)(stage[0] + sp * ROWP + sc);
  const unsigned lds1 = (unsigned)(uintptr_t)(stage[1] + sp * ROWP + sc);

  auto issue_tap = [&](int kp, unsigned ldsAddr) {
    const int ho = hs + kp / 3 - 1;
    const int wo = wsd + kp % 3 - 1;
    if (((unsigned)ho < (unsigned)HH) && ((unsigned)wo < (unsigned)WW)) {
      const unsigned char* g =
          actQ + ((size_t)(bs * HW + ho * WW + wo)) * CIN + sc;
      asm volatile("global_load_async_to_lds_b128 %0, %1, off"
                   :: "v"(ldsAddr), "v"(g) : "memory");
    }
  };

  issue_tap(0, lds0);
  asm volatile("s_wait_asynccnt 0" ::: "memory");
  __syncthreads();

  v8i acc0 = {}, acc1 = {};
  for (int kp = 0; kp < 9; ++kp) {
    const int cur = kp & 1;
    if (kp < 8) issue_tap(kp + 1, cur ? lds0 : lds1);

    const int hoc = hc + kp / 3 - 1;
    const int woc = wc + kp % 3 - 1;
    const bool bvalid =
        ((unsigned)hoc < (unsigned)HH) && ((unsigned)woc < (unsigned)WW);
    const unsigned char* bbase =
        (cur ? stage[1] : stage[0]) + lr * ROWP + hi * 16;
    const signed char* wk0 =
        wqs + ((((size_t)kp * 16 + mt0) * 4) * 32 + lane) * 32;
    const signed char* wk1 = wk0 + 4 * 32 * 32;

#pragma unroll
    for (int kci = 0; kci < 4; ++kci) {
      const int kc = kci * 64;
      const uint4* ap0 = (const uint4*)(wk0 + (size_t)kci * 32 * 32);
      const uint4* ap1 = (const uint4*)(wk1 + (size_t)kci * 32 * 32);
      uint4 q0 = ap0[0], q1 = ap0[1];
      uint4 r0 = ap1[0], r1 = ap1[1];
      v8i A0, A1;
      A0[0] = (int)q0.x; A0[1] = (int)q0.y; A0[2] = (int)q0.z; A0[3] = (int)q0.w;
      A0[4] = (int)q1.x; A0[5] = (int)q1.y; A0[6] = (int)q1.z; A0[7] = (int)q1.w;
      A1[0] = (int)r0.x; A1[1] = (int)r0.y; A1[2] = (int)r0.z; A1[3] = (int)r0.w;
      A1[4] = (int)r1.x; A1[5] = (int)r1.y; A1[6] = (int)r1.z; A1[7] = (int)r1.w;
      const unsigned char* plo = bvalid ? (bbase + kc)      : (const unsigned char*)zbuf;
      const unsigned char* phi = bvalid ? (bbase + kc + 32) : (const unsigned char*)zbuf;
      uint4 b0 = *(const uint4*)plo;
      uint4 b1 = *(const uint4*)phi;
      v8i Bv;
      Bv[0] = (int)b0.x; Bv[1] = (int)b0.y; Bv[2] = (int)b0.z; Bv[3] = (int)b0.w;
      Bv[4] = (int)b1.x; Bv[5] = (int)b1.y; Bv[6] = (int)b1.z; Bv[7] = (int)b1.w;
      acc0 = __builtin_amdgcn_wmma_i32_16x16x64_iu8(
          true, A0, false, Bv, acc0, false, false);
      acc1 = __builtin_amdgcn_wmma_i32_16x16x64_iu8(
          true, A1, false, Bv, acc1, false, false);
    }

    asm volatile("s_wait_asynccnt 0" ::: "memory");
    __syncthreads();
  }

  const size_t outpix = (size_t)bc * COUT * HW + (size_t)hc * WW + wc;
#pragma unroll
  for (int r = 0; r < 8; ++r) {
    const int oca = m0 + r + hi * 8;
    const int ocb = oca + 16;
    out[outpix + (size_t)oca * HW] = (float)acc0[r] * scale[oca] + bias[oca];
    out[outpix + (size_t)ocb * HW] = (float)acc1[r] * scale[ocb] + bias[ocb];
  }
}

extern "C" void kernel_launch(void* const* d_in, const int* in_sizes, int n_in,
                              void* d_out, int out_size, void* d_ws, size_t ws_size,
                              hipStream_t stream) {
  (void)in_sizes; (void)n_in; (void)out_size; (void)ws_size;
  const float* x     = (const float*)d_in[0];
  const float* gamma = (const float*)d_in[1];
  const float* beta  = (const float*)d_in[2];
  const float* a     = (const float*)d_in[3];
  const float* wfp   = (const float*)d_in[4];
  const float* bias  = (const float*)d_in[5];
  float* out = (float*)d_out;

  char* ws = (char*)d_ws;
  float*         mu    = (float*)(ws);
  float*         rs    = (float*)(ws + 1024);
  float*         scale = (float*)(ws + 2048);
  signed char*   wqs   = (signed char*)(ws + 4096);
  unsigned char* actQ  = (unsigned char*)(ws + (1 << 20));

  gn_stats_kernel<<<NB * 8, 256, 0, stream>>>(x, mu, rs);
  act_quant_kernel<<<(NPIX * 64) / 256, 256, 0, stream>>>(x, gamma, beta, a, mu, rs, actQ);
  wquant_kernel<<<COUT, 256, 0, stream>>>(wfp, a, wqs, scale);
  conv_gemm_kernel<<<NPIX / 16, 256, 0, stream>>>(actQ, wqs, scale, bias, out);
}
